// MultiStepPointer_68118181314642
// MI455X (gfx1250) — matrix operand order repaired and checked by reading, not compiled, not run on hardware
//
#include <hip/hip_runtime.h>

typedef _Float16 half_t;
typedef __attribute__((ext_vector_type(16))) _Float16 v16h;
typedef __attribute__((ext_vector_type(8)))  float    v8f;

#define LDP     72
#define NWAVES  4
#define MAXN    20

// LDS: f16 weight region (offsets in halfs)
#define OFF_NE   0        // ne_w zero-padded to [64][32]
#define OFF_M1A  2048     // mp1_w1 [64][128]
#define OFF_M1B  10240    // mp1_w2 [64][64]
#define OFF_M2A  14336    // mp2_w1 [64][128]
#define OFF_M2B  22528    // mp2_w2 [64][64]
#define OFF_PK   26624    // pk_w   [64][64]
#define HALF_TOT 30720

// f32 param region (offsets in floats)
#define PF_NEB   0
#define PF_M1B1  64
#define PF_M1B2  128
#define PF_N1G   192
#define PF_N1B   256
#define PF_M2B1  320
#define PF_M2B2  384
#define PF_N2G   448
#define PF_N2B   512
#define PF_PKB   576
#define PF_TOT   640

// per-wave f32 scratch (offsets in floats)
#define W_HT   0
#define W_TT   (32*LDP)
#define W_MSG  (2*32*LDP)
#define W_XV   (W_MSG+64)
#define W_RS   (W_XV+72)
#define W_NS   (W_RS+64)
#define W_Q    (W_NS+64)
#define W_TOT  (W_Q+64)

#define SMEM_BYTES (HALF_TOT*2 + (PF_TOT + NWAVES*W_TOT)*4)

struct KParams {
  const float* feats; const int* nn; const float* rnn; const float* onehot;
  const float* ne_w; const float* ne_b;
  const float* m1w1; const float* m1b1; const float* m1w2; const float* m1b2;
  const float* n1g;  const float* n1b;
  const float* m2w1; const float* m2b1; const float* m2w2; const float* m2b2;
  const float* n2g;  const float* n2b;
  const float* gwih; const float* gwhh; const float* gbih; const float* gbhh;
  const float* opw; const float* opb; const float* c1w; const float* c1b;
  const float* c2w; const float* c2b; const float* pqw; const float* pqb;
  const float* pkw; const float* pkb;
  float* out;
  int Btot;
};

__device__ __forceinline__ void wave_lds_sync() {
  // CDNA5 split wait counter: all prior DS ops complete; wave32 lockstep makes
  // this a full intra-wave LDS producer->consumer fence.
  asm volatile("s_wait_dscnt 0" ::: "memory");
}

__device__ __forceinline__ v8f wmma_f16(v16h a, v16h b, v8f c) {
  return __builtin_amdgcn_wmma_f32_16x16x32_f16(false, a, false, b, (short)0, c,
                                                false, false);
}

// A-fragment (16x32 f16) from f32 LDS tile rows [m*16 .. m*16+15]
__device__ __forceinline__ v16h a_frag_lds(const float* tile, int ktile, int lane) {
  const float* p = tile + (lane & 15) * LDP;
  const int kb = ktile * 32 + ((lane >> 4) << 3);
  v16h a;
#pragma unroll
  for (int i = 0; i < 8; ++i) a[i]     = (half_t)p[kb + i];
#pragma unroll
  for (int i = 0; i < 8; ++i) a[8 + i] = (half_t)p[kb + 16 + i];
  return a;
}

// A-fragment where every row is the same broadcast vector (msg concat half)
__device__ __forceinline__ v16h a_frag_bcast(const float* vec, int ktile, int lane) {
  const int kb = ktile * 32 + ((lane >> 4) << 3);
  v16h a;
#pragma unroll
  for (int i = 0; i < 8; ++i) a[i]     = (half_t)vec[kb + i];
#pragma unroll
  for (int i = 0; i < 8; ++i) a[8 + i] = (half_t)vec[kb + 16 + i];
  return a;
}

// A-fragment from global node_feats [20][16], K padded 16->32, rows>=20 zero
__device__ __forceinline__ v16h a_frag_feats(const float* feats, int m, int lane) {
  v16h a;
#pragma unroll
  for (int i = 0; i < 16; ++i) a[i] = (half_t)0.0f;
  const int row = m * 16 + (lane & 15);
  if (row < MAXN) {
    const float* p = feats + row * 16 + ((lane >> 4) << 3);
#pragma unroll
    for (int i = 0; i < 8; ++i) a[i] = (half_t)p[i];
  }
  return a;
}

// B-fragment (32x16 f16) from LDS f16 weight stored [N_total][K_total] row-major
__device__ __forceinline__ v16h b_frag(const half_t* W, int Ktot, int ntile,
                                       int ktile, int lane) {
  const half_t* p = W + (ntile * 16 + (lane & 15)) * Ktot
                      + ktile * 32 + ((lane >> 4) << 3);
  v16h b;
#pragma unroll
  for (int i = 0; i < 16; ++i) b[i] = p[i + ((i >> 3) << 3)];
  return b;
}

__device__ __forceinline__ void store_acc_bias(float* tile, int nt, int lane,
                                               v8f acc, float bb) {
  const int col = nt * 16 + (lane & 15);
  const int r0  = (lane >> 4) << 3;
#pragma unroll
  for (int v = 0; v < 8; ++v) tile[(r0 + v) * LDP + col] = acc[v] + bb;
}

__device__ __forceinline__ float sigm(float x) { return 1.0f / (1.0f + __expf(-x)); }

// masked mean-pool columns of hT -> msg[64]
__device__ __forceinline__ void pool_rows(const float* hT, float* msg, int lane,
                                          int nv, float inv_cnt) {
#pragma unroll
  for (int cc = 0; cc < 2; ++cc) {
    const int c = lane + 32 * cc;
    float s = 0.f;
    for (int r = 0; r < nv; ++r) s += hT[r * LDP + c];
    msg[c] = s * inv_cnt;
  }
  wave_lds_sync();
}

// one message-passing block: h = LN(h + W2*relu(W1*[h,msg]+b1)+b2)*mask
__device__ void mp_block(float* hT, float* tT, const float* msg,
                         const half_t* Wa, const half_t* Wb,
                         const float* b1, const float* b2,
                         const float* g, const float* bn, int lane, int nv) {
  v16h aM0 = a_frag_bcast(msg, 0, lane);
  v16h aM1 = a_frag_bcast(msg, 1, lane);
  // lin1 (K=128) + relu -> tT
  for (int m = 0; m < 2; ++m) {
    v16h a0 = a_frag_lds(hT + m * 16 * LDP, 0, lane);
    v16h a1 = a_frag_lds(hT + m * 16 * LDP, 1, lane);
#pragma unroll
    for (int nt = 0; nt < 4; ++nt) {
      v8f acc = {0.f,0.f,0.f,0.f,0.f,0.f,0.f,0.f};
      acc = wmma_f16(a0,  b_frag(Wa, 128, nt, 0, lane), acc);
      acc = wmma_f16(a1,  b_frag(Wa, 128, nt, 1, lane), acc);
      acc = wmma_f16(aM0, b_frag(Wa, 128, nt, 2, lane), acc);
      acc = wmma_f16(aM1, b_frag(Wa, 128, nt, 3, lane), acc);
      const int col = nt * 16 + (lane & 15);
      const int r0  = (lane >> 4) << 3;
      const float bb = b1[col];
#pragma unroll
      for (int v = 0; v < 8; ++v)
        tT[(m * 16 + r0 + v) * LDP + col] = fmaxf(acc[v] + bb, 0.0f);
    }
  }
  wave_lds_sync();
  // lin2 (K=64) + bias + residual -> tT (pre-norm)
  for (int m = 0; m < 2; ++m) {
    v16h a0 = a_frag_lds(tT + m * 16 * LDP, 0, lane);
    v16h a1 = a_frag_lds(tT + m * 16 * LDP, 1, lane);
#pragma unroll
    for (int nt = 0; nt < 4; ++nt) {
      v8f acc = {0.f,0.f,0.f,0.f,0.f,0.f,0.f,0.f};
      acc = wmma_f16(a0, b_frag(Wb, 64, nt, 0, lane), acc);
      acc = wmma_f16(a1, b_frag(Wb, 64, nt, 1, lane), acc);
      const int col = nt * 16 + (lane & 15);
      const int r0  = (lane >> 4) << 3;
      const float bb = b2[col];
#pragma unroll
      for (int v = 0; v < 8; ++v) {
        const int r = m * 16 + r0 + v;
        tT[r * LDP + col] = acc[v] + bb + hT[r * LDP + col];
      }
    }
  }
  wave_lds_sync();
  // LayerNorm over 64 features, lane == row, then mask
  {
    const int r = lane;
    const float* p = tT + r * LDP;
    float s = 0.f, ss = 0.f;
#pragma unroll 8
    for (int c = 0; c < 64; ++c) { float x = p[c]; s += x; ss += x * x; }
    const float mu  = s * 0.015625f;
    const float var = ss * 0.015625f - mu * mu;
    const float rstd = rsqrtf(var + 1e-5f);
    const float mf = (r < nv) ? 1.0f : 0.0f;
    float* q = hT + r * LDP;
#pragma unroll 8
    for (int c = 0; c < 64; ++c)
      q[c] = (((p[c] - mu) * rstd) * g[c] + bn[c]) * mf;
  }
  wave_lds_sync();
}

__global__ __launch_bounds__(NWAVES * 32)
void msp_kernel(KParams P) {
  extern __shared__ char smem[];
  half_t* sH = (half_t*)smem;
  float*  sP = (float*)(smem + HALF_TOT * 2);
  float*  sWv = sP + PF_TOT;

  const int tid = threadIdx.x;
  const int nthr = blockDim.x;

  // ---- stage weights (f32 -> f16) + params into LDS, once per block ----
  for (int i = tid; i < 2048; i += nthr) {
    const int o = i >> 5, k = i & 31;
    sH[OFF_NE + i] = (half_t)((k < 16) ? P.ne_w[o * 16 + k] : 0.0f);
  }
  for (int i = tid; i < 8192; i += nthr) sH[OFF_M1A + i] = (half_t)P.m1w1[i];
  for (int i = tid; i < 4096; i += nthr) sH[OFF_M1B + i] = (half_t)P.m1w2[i];
  for (int i = tid; i < 8192; i += nthr) sH[OFF_M2A + i] = (half_t)P.m2w1[i];
  for (int i = tid; i < 4096; i += nthr) sH[OFF_M2B + i] = (half_t)P.m2w2[i];
  for (int i = tid; i < 4096; i += nthr) sH[OFF_PK  + i] = (half_t)P.pkw[i];
  for (int i = tid; i < 64; i += nthr) {
    sP[PF_NEB  + i] = P.ne_b[i]; sP[PF_M1B1 + i] = P.m1b1[i];
    sP[PF_M1B2 + i] = P.m1b2[i]; sP[PF_N1G  + i] = P.n1g[i];
    sP[PF_N1B  + i] = P.n1b[i];  sP[PF_M2B1 + i] = P.m2b1[i];
    sP[PF_M2B2 + i] = P.m2b2[i]; sP[PF_N2G  + i] = P.n2g[i];
    sP[PF_N2B  + i] = P.n2b[i];  sP[PF_PKB  + i] = P.pkb[i];
  }
  __syncthreads();

  const int lane = tid & 31;
  const int wid  = tid >> 5;
  float* wsp = sWv + wid * W_TOT;
  float* hT  = wsp + W_HT;
  float* tT  = wsp + W_TT;
  float* msg = wsp + W_MSG;
  float* xv  = wsp + W_XV;
  float* rs  = wsp + W_RS;
  float* ns  = wsp + W_NS;
  float* qv  = wsp + W_Q;

  const long Bt = P.Btot;
  const long OUT_C1 = Bt * 8, OUT_C2 = Bt * 18, OUT_PTR = Bt * 28, OUT_ST = Bt * 48;
  const long stride = (long)gridDim.x * NWAVES;

  for (long b = (long)blockIdx.x * NWAVES + wid; b < Bt; b += stride) {
    const int nv = P.nn[b];
    const float inv_cnt = 1.0f / (float)(nv < 1 ? 1 : nv);
    if (b + stride < Bt)
      __builtin_prefetch(P.feats + (b + stride) * (MAXN * 16), 0, 1);

    // ---- node encoder: h = feats @ ne_w^T + ne_b ----
    for (int m = 0; m < 2; ++m) {
      v16h a = a_frag_feats(P.feats + b * (MAXN * 16), m, lane);
#pragma unroll
      for (int nt = 0; nt < 4; ++nt) {
        v8f acc = {0.f,0.f,0.f,0.f,0.f,0.f,0.f,0.f};
        acc = wmma_f16(a, b_frag(sH + OFF_NE, 32, nt, 0, lane), acc);
        store_acc_bias(hT + m * 16 * LDP, nt, lane, acc,
                       sP[PF_NEB + nt * 16 + (lane & 15)]);
      }
    }
    wave_lds_sync();

    pool_rows(hT, msg, lane, nv, inv_cnt);
    mp_block(hT, tT, msg, sH + OFF_M1A, sH + OFF_M1B,
             sP + PF_M1B1, sP + PF_M1B2, sP + PF_N1G, sP + PF_N1B, lane, nv);
    pool_rows(hT, msg, lane, nv, inv_cnt);
    mp_block(hT, tT, msg, sH + OFF_M2A, sH + OFF_M2B,
             sP + PF_M2B1, sP + PF_M2B2, sP + PF_N2G, sP + PF_N2B, lane, nv);
    pool_rows(hT, msg, lane, nv, inv_cnt);   // msg == graph_vec now

    // ---- GRU cell (VALU, lane-parallel over 192 gate rows) ----
    xv[lane] = msg[lane];
    xv[32 + lane] = msg[32 + lane];
    if (lane < 8) xv[64 + lane] = P.onehot[b * 8 + lane];
    rs[lane] = P.rnn[b * 64 + lane];
    rs[32 + lane] = P.rnn[b * 64 + 32 + lane];
    wave_lds_sync();

    float gi[6], gh[6];
#pragma unroll
    for (int j = 0; j < 6; ++j) {
      const int row = lane + 32 * j;
      const float* wi = P.gwih + row * 72;
      const float* wh = P.gwhh + row * 64;
      float si = P.gbih[row], sh = P.gbhh[row];
#pragma unroll 8
      for (int k = 0; k < 72; ++k) si = fmaf(xv[k], wi[k], si);
#pragma unroll 8
      for (int k = 0; k < 64; ++k) sh = fmaf(rs[k], wh[k], sh);
      gi[j] = si; gh[j] = sh;
    }
    float ns0, ns1;
    {
      const float r = sigm(gi[0] + gh[0]), z = sigm(gi[2] + gh[2]);
      const float n = tanhf(gi[4] + r * gh[4]);
      ns0 = (1.f - z) * n + z * rs[lane];
    }
    {
      const float r = sigm(gi[1] + gh[1]), z = sigm(gi[3] + gh[3]);
      const float n = tanhf(gi[5] + r * gh[5]);
      ns1 = (1.f - z) * n + z * rs[32 + lane];
    }
    ns[lane] = ns0; ns[32 + lane] = ns1;
    P.out[OUT_ST + b * 64 + lane] = ns0;
    P.out[OUT_ST + b * 64 + 32 + lane] = ns1;
    wave_lds_sync();

    // ---- heads: op(8) / c1(10) / c2(10), one output per lane ----
    if (lane < 28) {
      const float* w; float bias; long off;
      if (lane < 8)       { w = P.opw + lane * 64; bias = P.opb[lane]; off = b * 8 + lane; }
      else if (lane < 18) { const int o = lane - 8;  w = P.c1w + o * 64; bias = P.c1b[o]; off = OUT_C1 + b * 10 + o; }
      else                { const int o = lane - 18; w = P.c2w + o * 64; bias = P.c2b[o]; off = OUT_C2 + b * 10 + o; }
      float s = bias;
#pragma unroll 8
      for (int k = 0; k < 64; ++k) s = fmaf(ns[k], w[k], s);
      P.out[off] = s;
    }
    // pointer query: q = pq_w @ ns + pq_b
#pragma unroll
    for (int cc = 0; cc < 2; ++cc) {
      const int c = lane + 32 * cc;
      const float* w = P.pqw + c * 64;
      float s = P.pqb[c];
#pragma unroll 8
      for (int k = 0; k < 64; ++k) s = fmaf(ns[k], w[k], s);
      qv[c] = s;
    }
    wave_lds_sync();

    // ---- keys = h @ pk_w^T + pk_b (WMMA), then ptr logits ----
    for (int m = 0; m < 2; ++m) {
      v16h a0 = a_frag_lds(hT + m * 16 * LDP, 0, lane);
      v16h a1 = a_frag_lds(hT + m * 16 * LDP, 1, lane);
#pragma unroll
      for (int nt = 0; nt < 4; ++nt) {
        v8f acc = {0.f,0.f,0.f,0.f,0.f,0.f,0.f,0.f};
        acc = wmma_f16(a0, b_frag(sH + OFF_PK, 64, nt, 0, lane), acc);
        acc = wmma_f16(a1, b_frag(sH + OFF_PK, 64, nt, 1, lane), acc);
        store_acc_bias(tT + m * 16 * LDP, nt, lane, acc,
                       sP[PF_PKB + nt * 16 + (lane & 15)]);
      }
    }
    wave_lds_sync();
    if (lane < MAXN) {
      const float* kr = tT + lane * LDP;
      float s = 0.f;
#pragma unroll 8
      for (int k = 0; k < 64; ++k) s = fmaf(kr[k], qv[k], s);
      P.out[OUT_PTR + b * 20 + lane] = (lane < nv) ? s : -1e9f;
    }
    wave_lds_sync();
  }
}

extern "C" void kernel_launch(void* const* d_in, const int* in_sizes, int n_in,
                              void* d_out, int out_size, void* d_ws, size_t ws_size,
                              hipStream_t stream) {
  KParams P;
  P.feats  = (const float*)d_in[0];
  P.nn     = (const int*)  d_in[1];
  P.rnn    = (const float*)d_in[2];
  P.onehot = (const float*)d_in[3];
  P.ne_w = (const float*)d_in[4];  P.ne_b = (const float*)d_in[5];
  P.m1w1 = (const float*)d_in[6];  P.m1b1 = (const float*)d_in[7];
  P.m1w2 = (const float*)d_in[8];  P.m1b2 = (const float*)d_in[9];
  P.n1g  = (const float*)d_in[10]; P.n1b  = (const float*)d_in[11];
  P.m2w1 = (const float*)d_in[12]; P.m2b1 = (const float*)d_in[13];
  P.m2w2 = (const float*)d_in[14]; P.m2b2 = (const float*)d_in[15];
  P.n2g  = (const float*)d_in[16]; P.n2b  = (const float*)d_in[17];
  P.gwih = (const float*)d_in[18]; P.gwhh = (const float*)d_in[19];
  P.gbih = (const float*)d_in[20]; P.gbhh = (const float*)d_in[21];
  P.opw  = (const float*)d_in[22]; P.opb  = (const float*)d_in[23];
  P.c1w  = (const float*)d_in[24]; P.c1b  = (const float*)d_in[25];
  P.c2w  = (const float*)d_in[26]; P.c2b  = (const float*)d_in[27];
  P.pqw  = (const float*)d_in[28]; P.pqb  = (const float*)d_in[29];
  P.pkw  = (const float*)d_in[30]; P.pkb  = (const float*)d_in[31];
  P.out  = (float*)d_out;
  P.Btot = in_sizes[1];   // element count of n_nodes == B

  int grid = 1024;
  if ((long)grid * NWAVES > P.Btot) grid = (P.Btot + NWAVES - 1) / NWAVES;
  msp_kernel<<<dim3(grid), dim3(NWAVES * 32), SMEM_BYTES, stream>>>(P);
}
